// ConformalGQA_73924977098966
// MI455X (gfx1250) — matrix operand order repaired and checked by reading, not compiled, not run on hardware
//
#include <hip/hip_runtime.h>
#include <hip/hip_bf16.h>

typedef __bf16 bh;
typedef __attribute__((ext_vector_type(16))) __bf16 v16bf;
typedef __attribute__((ext_vector_type(8)))  float   v8f;

static constexpr int BATCH  = 4;
static constexpr int SEQ    = 1024;
static constexpr int DMODEL = 2048;
static constexpr int DKV    = 512;
static constexpr int NHEADS = 32;
static constexpr int NKVH   = 8;
static constexpr int HDIM   = 64;

static constexpr int LDS_STRIDE = 72;  // 64 halves + 8 pad (144B rows, 16B aligned)

union Frag { v16bf v; uint4 u[2]; };

static __device__ __forceinline__ v8f wmma_bf16(v16bf a, v16bf b, v8f c) {
  return __builtin_amdgcn_wmma_f32_16x16x32_bf16(false, a, false, b, (short)0, c, false, false);
}

// ---------------------------------------------------------------------------
// Tiled bf16 GEMM with fused epilogues.
//   C[M,N] = A[M,K] (bf16 row-major) * Bt[N,K]^T (bf16)
// block = 256 threads (8 waves), 128x128 C tile, K staged 64 at a time in LDS
// (two 32-deep WMMA steps per barrier; 16 WMMAs per stage per wave).
// EPI 0: store f32 C.           EPI 1: RoPE -> Qb (b,H,t,hd) bf16.
// EPI 2: RoPE + 0.5|k|^2 -> Kb. EPI 3: transpose -> Vt (b,KVH,hd,t) bf16.
// ---------------------------------------------------------------------------
template <int EPI>
__global__ __launch_bounds__(256)
void gemm_epi(const bh* __restrict__ A, const bh* __restrict__ Bt,
              float* __restrict__ Cf, bh* __restrict__ Cb,
              float* __restrict__ kn, int M, int N, int K) {
  __shared__ bh As[128 * LDS_STRIDE];
  __shared__ bh Bs[128 * LDS_STRIDE];
  const int tid   = threadIdx.x;
  const int lane  = tid & 31;
  const int w     = tid >> 5;
  const int lhalf = lane & 15;
  const int lsel  = lane >> 4;
  const int wm    = (w & 3) * 32;   // wave row base within block tile
  const int wn    = (w >> 2) * 64;  // wave col base within block tile (one head)
  const int m0    = blockIdx.y * 128;
  const int n0    = blockIdx.x * 128;

  v8f acc[2][4];
#pragma unroll
  for (int i = 0; i < 2; i++)
#pragma unroll
    for (int j = 0; j < 4; j++)
#pragma unroll
      for (int r = 0; r < 8; r++) acc[i][j][r] = 0.0f;

  const int srow = tid >> 1;         // 0..127
  const int scol = (tid & 1) * 32;   // 0 or 32 (halves)

  for (int k0 = 0; k0 < K; k0 += 64) {
    // stage 128x64 tiles of A and Bt into LDS (each thread: 32 halves of each)
    const uint4* ga = (const uint4*)(A  + (size_t)(m0 + srow) * K + k0 + scol);
    const uint4* gb = (const uint4*)(Bt + (size_t)(n0 + srow) * K + k0 + scol);
    uint4 a0 = ga[0], a1 = ga[1], a2 = ga[2], a3 = ga[3];
    uint4 b0 = gb[0], b1 = gb[1], b2 = gb[2], b3 = gb[3];
    if (k0 + 64 < K) {  // prefetch next k-tile
      __builtin_prefetch(A  + (size_t)(m0 + srow) * K + k0 + 64 + scol, 0, 3);
      __builtin_prefetch(Bt + (size_t)(n0 + srow) * K + k0 + 64 + scol, 0, 3);
    }
    uint4* la = (uint4*)(As + srow * LDS_STRIDE + scol);
    uint4* lb = (uint4*)(Bs + srow * LDS_STRIDE + scol);
    la[0] = a0; la[1] = a1; la[2] = a2; la[3] = a3;
    lb[0] = b0; lb[1] = b1; lb[2] = b2; lb[3] = b3;
    __syncthreads();

#pragma unroll
    for (int ks = 0; ks < 64; ks += 32) {
      // A fragments: lane = row M=lhalf; K = ks+16*lsel .. +15 contiguous
      Frag af[2];
#pragma unroll
      for (int mi = 0; mi < 2; mi++) {
        const bh* base = As + (wm + mi * 16 + lhalf) * LDS_STRIDE + ks;
        af[mi].u[0] = *(const uint4*)(base + 16 * lsel);
        af[mi].u[1] = *(const uint4*)(base + 16 * lsel + 8);
      }
      // B fragments: lane = col N=lhalf; K = ks+16*lsel .. +15 contiguous
      v16bf bfr[4];
#pragma unroll
      for (int ni = 0; ni < 4; ni++)
        bfr[ni] = *(const v16bf*)(Bs + (wn + ni * 16 + lhalf) * LDS_STRIDE +
                                  ks + 16 * lsel);
#pragma unroll
      for (int mi = 0; mi < 2; mi++)
#pragma unroll
        for (int ni = 0; ni < 4; ni++)
          acc[mi][ni] = wmma_bf16(af[mi].v, bfr[ni], acc[mi][ni]);
    }
    __syncthreads();
  }

  // ---------------- epilogues ----------------
  if (EPI == 0) {
#pragma unroll
    for (int mi = 0; mi < 2; mi++)
#pragma unroll
      for (int ni = 0; ni < 4; ni++) {
        const int col = n0 + wn + ni * 16 + lhalf;
#pragma unroll
        for (int r = 0; r < 8; r++) {
          const int row = m0 + wm + mi * 16 + r + 8 * lsel;
          Cf[(size_t)row * N + col] = acc[mi][ni][r];
        }
      }
  } else if (EPI == 1 || EPI == 2) {
    // RoPE: lane holds cols j = ni*16+lhalf (ni=0,1) paired with j+32 (ni+2).
    const int h = (n0 + wn) >> 6;  // head index within this projection
    const int nh = (EPI == 1) ? NHEADS : NKVH;
#pragma unroll
    for (int mi = 0; mi < 2; mi++) {
#pragma unroll
      for (int r = 0; r < 8; r++) {
        const int row = m0 + wm + mi * 16 + r + 8 * lsel;
        const int bb = row >> 10;
        const int t  = row & (SEQ - 1);
        bh* dst = Cb + ((size_t)(bb * nh + h) * SEQ + t) * HDIM;
        float ss = 0.0f;
#pragma unroll
        for (int ni = 0; ni < 2; ni++) {
          const int j = ni * 16 + lhalf;  // 0..31
          const float freq = __powf(10000.0f, -(float)j * (1.0f / 32.0f));
          float s, c;
          __sincosf((float)t * freq, &s, &c);
          const float q1 = acc[mi][ni][r];
          const float q2 = acc[mi][ni + 2][r];
          const float o1 = q1 * c - q2 * s;
          const float o2 = q2 * c + q1 * s;
          dst[j]      = (bh)o1;
          dst[j + 32] = (bh)o2;
          if (EPI == 2) ss += o1 * o1 + o2 * o2;
        }
        if (EPI == 2) {  // reduce 0.5*|k|^2 across the 16-lane half-wave
          ss += __shfl_xor(ss, 1, 16);
          ss += __shfl_xor(ss, 2, 16);
          ss += __shfl_xor(ss, 4, 16);
          ss += __shfl_xor(ss, 8, 16);
          if (lhalf == 0) kn[(size_t)(bb * NKVH + h) * SEQ + t] = 0.5f * ss;
        }
      }
    }
  } else {  // EPI == 3: V transpose -> (b, KVH, hd, t) bf16
    const int h = (n0 + wn) >> 6;
#pragma unroll
    for (int mi = 0; mi < 2; mi++)
#pragma unroll
      for (int ni = 0; ni < 4; ni++) {
        const int j = ni * 16 + lhalf;
#pragma unroll
        for (int r = 0; r < 8; r++) {
          const int row = m0 + wm + mi * 16 + r + 8 * lsel;
          const int bb = row >> 10;
          const int t  = row & (SEQ - 1);
          Cb[((size_t)(bb * NKVH + h) * HDIM + j) * SEQ + t] = (bh)acc[mi][ni][r];
        }
      }
  }
}

// ---------------------------------------------------------------------------
// Layout kernels
// ---------------------------------------------------------------------------
__global__ void cvt_bf16_kernel(const float* __restrict__ src, bh* __restrict__ dst, int n) {
  int i = blockIdx.x * blockDim.x + threadIdx.x;
  if (i < n) dst[i] = (bh)src[i];
}

// W[K,N] f32 -> Wt[N,K] bf16
__global__ void wtr_kernel(const float* __restrict__ W, bh* __restrict__ Wt, int K, int N) {
  int i = blockIdx.x * blockDim.x + threadIdx.x;
  if (i >= K * N) return;
  int k = i / N, n = i % N;
  Wt[(size_t)n * K + k] = (bh)W[i];
}

// ---------------------------------------------------------------------------
// Flash-style causal GQA attention. One wave per (b, h, 16 q-rows).
// scores = (q.k - 0.5|k|^2)/8  (q-norm term is row-constant: dropped, exact)
// ---------------------------------------------------------------------------
__global__ __launch_bounds__(128)
void attn_kernel(const bh* __restrict__ Qb, const bh* __restrict__ Kb,
                 const bh* __restrict__ Vt, const float* __restrict__ kn,
                 bh* __restrict__ Yb) {
  __shared__ bh Pl[4][16 * 32];
  const int lane  = threadIdx.x & 31;
  const int w     = threadIdx.x >> 5;
  const int lhalf = lane & 15;
  const int lsel  = lane >> 4;
  const int qbase = (blockIdx.x * 4 + w) * 16;
  const int h     = blockIdx.y;
  const int bb    = blockIdx.z;
  const int kvh   = h >> 2;  // GQA: 4 query heads per kv head

  // Q fragments: 16x64 reduction = two 16x32 A-frags
  Frag aq[2];
  const bh* qrow = Qb + ((size_t)(bb * NHEADS + h) * SEQ + qbase + lhalf) * HDIM;
#pragma unroll
  for (int f = 0; f < 2; f++) {
    aq[f].u[0] = *(const uint4*)(qrow + 32 * f + 16 * lsel);
    aq[f].u[1] = *(const uint4*)(qrow + 32 * f + 16 * lsel + 8);
  }

  float rowmax[8], rowsum[8];
  v8f acc[4];
#pragma unroll
  for (int r = 0; r < 8; r++) { rowmax[r] = -1e30f; rowsum[r] = 0.0f; }
#pragma unroll
  for (int ni = 0; ni < 4; ni++)
#pragma unroll
    for (int r = 0; r < 8; r++) acc[ni][r] = 0.0f;

  const bh*    kbase = Kb + (size_t)(bb * NKVH + kvh) * SEQ * HDIM;
  const bh*    vbase = Vt + (size_t)(bb * NKVH + kvh) * HDIM * SEQ;
  const float* knb   = kn + (size_t)(bb * NKVH + kvh) * SEQ;

  for (int kt0 = 0; kt0 < qbase + 16; kt0 += 32) {
    // S = Q K^T for 32 k-positions (two 16x16 tiles), reduction over hd=64
    v8f s0, s1;
#pragma unroll
    for (int r = 0; r < 8; r++) { s0[r] = 0.0f; s1[r] = 0.0f; }
#pragma unroll
    for (int f = 0; f < 2; f++) {
      const bh* kp0 = kbase + (size_t)(kt0 + lhalf) * HDIM + 32 * f + 16 * lsel;
      const bh* kp1 = kbase + (size_t)(kt0 + 16 + lhalf) * HDIM + 32 * f + 16 * lsel;
      v16bf bk0 = *(const v16bf*)kp0;
      v16bf bk1 = *(const v16bf*)kp1;
      s0 = wmma_bf16(aq[f].v, bk0, s0);
      s1 = wmma_bf16(aq[f].v, bk1, s1);
    }

    const float kn0 = knb[kt0 + lhalf];
    const float kn1 = knb[kt0 + 16 + lhalf];

    float p0[8], p1[8], sc[8];
#pragma unroll
    for (int r = 0; r < 8; r++) {
      const int qm = qbase + r + 8 * lsel;
      float v0 = (s0[r] - kn0) * 0.125f;
      float v1 = (s1[r] - kn1) * 0.125f;
      if (kt0 + lhalf > qm)      v0 = -1e30f;   // causal mask
      if (kt0 + 16 + lhalf > qm) v1 = -1e30f;
      float cm = fmaxf(v0, v1);
      cm = fmaxf(cm, __shfl_xor(cm, 1, 16));
      cm = fmaxf(cm, __shfl_xor(cm, 2, 16));
      cm = fmaxf(cm, __shfl_xor(cm, 4, 16));
      cm = fmaxf(cm, __shfl_xor(cm, 8, 16));
      const float nm    = fmaxf(rowmax[r], cm);
      const float scale = __expf(rowmax[r] - nm);
      rowmax[r] = nm;
      const float e0 = __expf(v0 - nm);
      const float e1 = __expf(v1 - nm);
      float rs = e0 + e1;
      rs += __shfl_xor(rs, 1, 16);
      rs += __shfl_xor(rs, 2, 16);
      rs += __shfl_xor(rs, 4, 16);
      rs += __shfl_xor(rs, 8, 16);
      rowsum[r] = rowsum[r] * scale + rs;
      sc[r] = scale; p0[r] = e0; p1[r] = e1;
    }
#pragma unroll
    for (int ni = 0; ni < 4; ni++)
#pragma unroll
      for (int r = 0; r < 8; r++) acc[ni][r] *= sc[r];

    // transpose P from C-layout to A-layout through per-wave LDS
    bh* pw = Pl[w];
#pragma unroll
    for (int r = 0; r < 8; r++) {
      pw[(r + 8 * lsel) * 32 + lhalf]      = (bh)p0[r];
      pw[(r + 8 * lsel) * 32 + 16 + lhalf] = (bh)p1[r];
    }
    asm volatile("" ::: "memory");  // DS ops are in-order per wave
    Frag pa;
    {
      const bh* pr = pw + lhalf * 32;
      pa.u[0] = *(const uint4*)(pr + 16 * lsel);
      pa.u[1] = *(const uint4*)(pr + 16 * lsel + 8);
    }

    // O += P (16x32) * V (32x64): V rows contiguous in t thanks to transpose
#pragma unroll
    for (int ni = 0; ni < 4; ni++) {
      const bh* vp = vbase + (size_t)(16 * ni + lhalf) * SEQ + kt0 + 16 * lsel;
      v16bf bv = *(const v16bf*)vp;
      acc[ni] = wmma_bf16(pa.v, bv, acc[ni]);
    }
  }

  // normalize and store Y in (b, t, H*hd) bf16 layout for the out-projection
#pragma unroll
  for (int ni = 0; ni < 4; ni++)
#pragma unroll
    for (int r = 0; r < 8; r++) {
      const int row = qbase + r + 8 * lsel;
      const float o = acc[ni][r] / rowsum[r];
      Yb[((size_t)bb * SEQ + row) * DMODEL + h * HDIM + 16 * ni + lhalf] = (bh)o;
    }
}

// ---------------------------------------------------------------------------
extern "C" void kernel_launch(void* const* d_in, const int* in_sizes, int n_in,
                              void* d_out, int out_size, void* d_ws, size_t ws_size,
                              hipStream_t stream) {
  const float* x  = (const float*)d_in[0];
  const float* Wq = (const float*)d_in[1];
  const float* Wk = (const float*)d_in[2];
  const float* Wv = (const float*)d_in[3];
  const float* Wo = (const float*)d_in[4];
  float* out = (float*)d_out;

  char* p = (char*)d_ws;
  auto alloc = [&](size_t bytes) {
    char* r = p;
    p += (bytes + 255) & ~(size_t)255;
    return r;
  };
  bh*    xb  = (bh*)   alloc((size_t)BATCH * SEQ * DMODEL * 2);
  bh*    Wqt = (bh*)   alloc((size_t)DMODEL * DMODEL * 2);
  bh*    Wkt = (bh*)   alloc((size_t)DKV * DMODEL * 2);
  bh*    Wvt = (bh*)   alloc((size_t)DKV * DMODEL * 2);
  bh*    Wot = (bh*)   alloc((size_t)DMODEL * DMODEL * 2);
  bh*    Qb  = (bh*)   alloc((size_t)BATCH * NHEADS * SEQ * HDIM * 2);
  bh*    Kb  = (bh*)   alloc((size_t)BATCH * NKVH * SEQ * HDIM * 2);
  bh*    Vt  = (bh*)   alloc((size_t)BATCH * NKVH * HDIM * SEQ * 2);
  float* kn  = (float*)alloc((size_t)BATCH * NKVH * SEQ * 4);
  bh*    Yb  = (bh*)   alloc((size_t)BATCH * SEQ * DMODEL * 2);

  int n;
  n = BATCH * SEQ * DMODEL;
  cvt_bf16_kernel<<<(n + 255) / 256, 256, 0, stream>>>(x, xb, n);
  n = DMODEL * DMODEL;
  wtr_kernel<<<(n + 255) / 256, 256, 0, stream>>>(Wq, Wqt, DMODEL, DMODEL);
  n = DMODEL * DKV;
  wtr_kernel<<<(n + 255) / 256, 256, 0, stream>>>(Wk, Wkt, DMODEL, DKV);
  wtr_kernel<<<(n + 255) / 256, 256, 0, stream>>>(Wv, Wvt, DMODEL, DKV);
  n = DMODEL * DMODEL;
  wtr_kernel<<<(n + 255) / 256, 256, 0, stream>>>(Wo, Wot, DMODEL, DMODEL);

  const int M = BATCH * SEQ;
  // Q projection with fused RoPE -> Qb (b,H,t,hd)
  gemm_epi<1><<<dim3(DMODEL / 128, M / 128), 256, 0, stream>>>(
      xb, Wqt, nullptr, Qb, nullptr, M, DMODEL, DMODEL);
  // K projection with fused RoPE + 0.5|k|^2 -> Kb, kn
  gemm_epi<2><<<dim3(DKV / 128, M / 128), 256, 0, stream>>>(
      xb, Wkt, nullptr, Kb, kn, M, DKV, DMODEL);
  // V projection with fused transpose -> Vt (b,KVH,hd,t)
  gemm_epi<3><<<dim3(DKV / 128, M / 128), 256, 0, stream>>>(
      xb, Wvt, nullptr, Vt, nullptr, M, DKV, DMODEL);

  attn_kernel<<<dim3(SEQ / 64, NHEADS, BATCH), 128, 0, stream>>>(Qb, Kb, Vt, kn, Yb);

  // output projection -> f32 d_out
  gemm_epi<0><<<dim3(DMODEL / 128, M / 128), 256, 0, stream>>>(
      Yb, Wot, out, nullptr, nullptr, M, DMODEL, DMODEL);
}
